// GCNBertClassifier_81398220194210
// MI455X (gfx1250) — compile-verified, matrix operand order checked by reading, not run on hardware
//
#include <hip/hip_runtime.h>

#define CB 32
#define CL 256
#define CD 768
#define CH 12
#define CDK 64
#define CMEM 384
#define CPOL 3
#define LDW 260
#define BIGMOD 0x40000000

typedef __attribute__((ext_vector_type(16))) __bf16 v16bf;
typedef __attribute__((ext_vector_type(8)))  float  v8f;
typedef unsigned short u16;
typedef long long i64;

struct alignas(16) U128 { unsigned int a, b, c, d; };
union FragU { v16bf v; U128 q[2]; };

__device__ __forceinline__ u16 f2bf(float f) {
    unsigned int u = __float_as_uint(f);
    u += 0x7FFFu + ((u >> 16) & 1u);
    return (u16)(u >> 16);
}

// NT fragment load: A is (M,K) row-major, B2 is (N,K) row-major (i.e. B^T).
// 16-bit WMMA fragment: lanes 0-15 hold rows r0..r0+15 with K-chunks [k0..k0+7],[k0+16..k0+23];
// lanes 16-31 hold the same rows with K-chunks [k0+8..k0+15],[k0+24..k0+31].
__device__ __forceinline__ v16bf load_frag(const u16* base, int ld, int r0, int k0, int lane) {
    int r  = r0 + (lane & 15);
    int kb = k0 + ((lane >> 4) << 3);
    const u16* p = base + (i64)r * ld + kb;
    FragU f;
    f.q[0] = *(const U128*)(p);
    f.q[1] = *(const U128*)(p + 16);
    return f.v;
}

__device__ __forceinline__ float blk_sum(float v, float* red) {
    int t = threadIdx.x;
    red[t] = v; __syncthreads();
    for (int s = 128; s > 0; s >>= 1) { if (t < s) red[t] += red[t + s]; __syncthreads(); }
    float r = red[0]; __syncthreads(); return r;
}
__device__ __forceinline__ float blk_max(float v, float* red) {
    int t = threadIdx.x;
    red[t] = v; __syncthreads();
    for (int s = 128; s > 0; s >>= 1) { if (t < s) red[t] = fmaxf(red[t], red[t + s]); __syncthreads(); }
    float r = red[0]; __syncthreads(); return r;
}

// ---------------------------------------------------------------- utilities
__global__ void k_zero(float* p, i64 n) {
    i64 i  = (i64)blockIdx.x * blockDim.x + threadIdx.x;
    i64 st = (i64)gridDim.x * blockDim.x;
    for (; i < n; i += st) p[i] = 0.f;
}

// (R,C) f32 row-major -> (C,R) bf16 row-major, batched over blockIdx.z
__global__ void k_transpose(const float* __restrict__ in, u16* __restrict__ out, int R, int C) {
    i64 base = (i64)blockIdx.z * R * C;
    int i = blockIdx.x * 256 + threadIdx.x;
    if (i >= R * C) return;
    int r = i / C, c = i % C;
    out[base + (i64)c * R + r] = f2bf(in[base + i]);
}

// ---------------------------------------------------------------- layernorm
__global__ void k_layernorm(const float* __restrict__ seq, const float* __restrict__ g,
                            const float* __restrict__ bta, float* __restrict__ xf,
                            u16* __restrict__ xb) {
    __shared__ float red[256];
    int row = blockIdx.x, tid = threadIdx.x;
    const float* p = seq + (i64)row * CD;
    float v0 = p[tid], v1 = p[tid + 256], v2 = p[tid + 512];
    float s  = blk_sum(v0 + v1 + v2, red);
    float sq = blk_sum(v0 * v0 + v1 * v1 + v2 * v2, red);
    float mu  = s * (1.f / CD);
    float var = (sq - (float)CD * mu * mu) * (1.f / (CD - 1));   // ddof = 1
    float inv = 1.f / (sqrtf(fmaxf(var, 0.f)) + 1e-6f);
#pragma unroll
    for (int i = 0; i < 3; ++i) {
        int d = tid + i * 256;
        float xv = g[d] * (p[d] - mu) * inv + bta[d];
        xf[(i64)row * CD + d] = xv;
        xb[(i64)row * CD + d] = f2bf(xv);
    }
}

// ---------------------------------------------------------------- generic NT WMMA GEMM
// C[z] = A[(z/aDiv)%aMod] (M,K) @ B2[(z/bDiv)%bMod]^T (N,K)  [+bias][/rowdiv][relu]
// Register blocking: each wave computes a 16x64 strip (1 A fragment reused by 4 B fragments,
// 4 back-to-back WMMAs per K step). N must be a multiple of 64, M a multiple of 64.
__global__ void __launch_bounds__(128)
k_gemm_nt(const u16* __restrict__ A, i64 sA, int aDiv, int aMod,
          const u16* __restrict__ Bt, i64 sB, int bDiv, int bMod,
          float* Cf, u16* Cb, i64 sC,
          int M, int N, int K, int lda, int ldb, int ldc,
          const float* bias, const float* rowdiv, i64 rdStride, int relu) {
    int z = blockIdx.z;
    const u16* Ab = A + (i64)((z / aDiv) % aMod) * sA;
    const u16* Bb = Bt + (i64)((z / bDiv) % bMod) * sB;
    i64 cOff = (i64)z * sC;
    int wave = threadIdx.x >> 5;
    int lane = threadIdx.x & 31;
    int tm  = blockIdx.y * 4 + wave;     // 16-row tile index
    int tn0 = blockIdx.x * 4;            // first of four 16-col tiles
    if (tm * 16 >= M) return;
    v8f acc0 = {}, acc1 = {}, acc2 = {}, acc3 = {};
    for (int k0 = 0; k0 < K; k0 += 32) {
        // speculative prefetch of the K-chunk two iterations ahead (dropped if OOB)
        {
            int kp = k0 + 64;
            const u16* pa = Ab + (i64)(tm * 16 + (lane & 15)) * lda + kp;
            const u16* pb = Bb + (i64)(tn0 * 16 + (lane & 15)) * ldb + kp;
            __builtin_prefetch(pa, 0, 1);
            __builtin_prefetch(pb, 0, 1);
        }
        v16bf af  = load_frag(Ab, lda, tm * 16, k0, lane);
        v16bf bf0 = load_frag(Bb, ldb, (tn0 + 0) * 16, k0, lane);
        v16bf bf1 = load_frag(Bb, ldb, (tn0 + 1) * 16, k0, lane);
        v16bf bf2 = load_frag(Bb, ldb, (tn0 + 2) * 16, k0, lane);
        v16bf bf3 = load_frag(Bb, ldb, (tn0 + 3) * 16, k0, lane);
        acc0 = __builtin_amdgcn_wmma_f32_16x16x32_bf16(false, af, false, bf0, (short)0, acc0, false, false);
        acc1 = __builtin_amdgcn_wmma_f32_16x16x32_bf16(false, af, false, bf1, (short)0, acc1, false, false);
        acc2 = __builtin_amdgcn_wmma_f32_16x16x32_bf16(false, af, false, bf2, (short)0, acc2, false, false);
        acc3 = __builtin_amdgcn_wmma_f32_16x16x32_bf16(false, af, false, bf3, (short)0, acc3, false, false);
    }
    int rb = tm * 16 + ((lane >> 4) << 3);
    const float* rd = rowdiv ? rowdiv + (i64)z * rdStride : nullptr;
    v8f accs[4] = { acc0, acc1, acc2, acc3 };
#pragma unroll
    for (int j = 0; j < 4; ++j) {
        int c = (tn0 + j) * 16 + (lane & 15);
        float bv = bias ? bias[c] : 0.f;
#pragma unroll
        for (int g = 0; g < 8; ++g) {
            int r = rb + g;
            float v = accs[j][g] + bv;
            if (rd) v /= rd[r];
            if (relu) v = fmaxf(v, 0.f);
            i64 idx = cOff + (i64)r * ldc + c;
            if (Cf) Cf[idx] = v;
            if (Cb) Cb[idx] = f2bf(v);
        }
    }
}

// ---------------------------------------------------------------- asp fix: asp = mask*(x@wd) + bd
__global__ void k_asp_fix(const float* __restrict__ t, const float* __restrict__ amask,
                          const float* __restrict__ bd, u16* __restrict__ out) {
    int i = blockIdx.x * 256 + threadIdx.x;
    if (i >= CB * CL * CDK) return;
    int row = i / CDK, c = i % CDK;
    out[i] = f2bf(amask[row] * t[i] + bd[c]);
}

// ---------------------------------------------------------------- fused asps: per (b,h) 256x256 tile in LDS
// asps = tanh(aspm@k_h^T + bias); avg = colmean; aspect_scores select; accumulate mean over h.
__global__ void k_asps_fused(const u16* __restrict__ aspm, const u16* __restrict__ kb,
                             const float* __restrict__ amask, const float* __restrict__ bm,
                             float* __restrict__ asa) {
    extern __shared__ float S[];
    float* avg = S + 256 * LDW;
    int z = blockIdx.x, b = z / CH, h = z % CH;
    int tid = threadIdx.x, wave = tid >> 5, lane = tid & 31;
    const u16* abase = aspm + (i64)z * CL * CDK;
    const u16* kbase = kb + (i64)b * CL * CD + h * CDK;
    float bias0 = bm[0];
    for (int t = wave; t < 256; t += 8) {
        int tm = t >> 4, tn = t & 15;
        v8f acc = {};
        for (int k0 = 0; k0 < CDK; k0 += 32) {
            v16bf af = load_frag(abase, CDK, tm * 16, k0, lane);
            v16bf bf = load_frag(kbase, CD,  tn * 16, k0, lane);
            acc = __builtin_amdgcn_wmma_f32_16x16x32_bf16(false, af, false, bf, (short)0, acc, false, false);
        }
        int c  = tn * 16 + (lane & 15);
        int rb = tm * 16 + ((lane >> 4) << 3);
#pragma unroll
        for (int g = 0; g < 8; ++g)
            S[(rb + g) * LDW + c] = tanhf(acc[g] + bias0);
    }
    __syncthreads();
    { float s = 0.f; for (int l = 0; l < CL; ++l) s += S[l * LDW + tid]; avg[tid] = s * (1.f / CL); }
    __syncthreads();
    const float invH = 1.f / CH;
    const float* am = amask + b * CL;
    for (int idx = tid; idx < CL * CL; idx += 256) {
        int l = idx >> 8, m = idx & 255;
        bool rl = am[l] > 0.f, cm = am[m] > 0.f;
        float ai = avg[l], aj = avg[m], v;
        if (rl && cm)     v = (l >= m) ? aj : ai;
        else if (rl)      v = aj;
        else if (cm)      v = ai;
        else              v = S[l * LDW + m];
        atomicAdd(&asa[(i64)b * CL * CL + idx], v * invH);
    }
}

// ---------------------------------------------------------------- fused attention: scores+softmax+head-mean
__global__ void k_attn_fused(const u16* __restrict__ qb, const u16* __restrict__ kb,
                             const float* __restrict__ smask, float* __restrict__ pre) {
    extern __shared__ float S[];
    int z = blockIdx.x, b = z / CH, h = z % CH;
    int tid = threadIdx.x, wave = tid >> 5, lane = tid & 31;
    const u16* qbase = qb + (i64)b * CL * CD + h * CDK;
    const u16* kbase = kb + (i64)b * CL * CD + h * CDK;
    for (int t = wave; t < 256; t += 8) {
        int tm = t >> 4, tn = t & 15;
        v8f acc = {};
        for (int k0 = 0; k0 < CDK; k0 += 32) {
            v16bf af = load_frag(qbase, CD, tm * 16, k0, lane);
            v16bf bf = load_frag(kbase, CD, tn * 16, k0, lane);
            acc = __builtin_amdgcn_wmma_f32_16x16x32_bf16(false, af, false, bf, (short)0, acc, false, false);
        }
        int c  = tn * 16 + (lane & 15);
        int rb = tm * 16 + ((lane >> 4) << 3);
        float mcol = smask[b * CL + c];
#pragma unroll
        for (int g = 0; g < 8; ++g) {
            float v = acc[g] * 0.125f;           // 1/sqrt(64)
            if (mcol == 0.f) v = -1e9f;
            S[(rb + g) * LDW + c] = v;
        }
    }
    __syncthreads();
    {   // softmax of row `tid`
        float* row = &S[tid * LDW];
        float mx = -3.4e38f;
        for (int j = 0; j < CL; ++j) mx = fmaxf(mx, row[j]);
        float s = 0.f;
        for (int j = 0; j < CL; ++j) { float e = __expf(row[j] - mx); row[j] = e; s += e; }
        float inv = 1.f / s;
        for (int j = 0; j < CL; ++j) row[j] *= inv;
    }
    __syncthreads();
    const float invH = 1.f / CH;
    for (int idx = tid; idx < CL * CL; idx += 256) {
        int r = idx >> 8, c = idx & 255;
        atomicAdd(&pre[(i64)b * CL * CL + idx], S[r * LDW + c] * invH);
    }
}

// ---------------------------------------------------------------- adjacency build
__global__ void k_adj_build(const float* __restrict__ pre, const float* __restrict__ asa,
                            const float* __restrict__ adjr, const float* __restrict__ smask,
                            float* adj_s_f, u16* adj_s_b, float* adj_ag_f, u16* adj_ag_b) {
    int i = blockIdx.x * 256 + threadIdx.x;
    if (i >= CB * CL * CL) return;
    int b = i / (CL * CL), rem = i % (CL * CL), l = rem / CL, m = rem % CL;
    float vs = (l == m) ? 1.f : pre[i];
    vs *= smask[b * CL + l];
    adj_s_f[i] = vs; adj_s_b[i] = f2bf(vs);
    float a = asa[i];
    float r = (a > 0.25f) ? 1.f : __expf(1.0f * adjr[i]);   // BETA=0.25, ALPHA=1.0
    float ag = r * a;
    adj_ag_f[i] = ag; adj_ag_b[i] = f2bf(ag);
}

// ---------------------------------------------------------------- denominators + KL
__global__ void k_rowstats(const float* __restrict__ adj_ag, const float* __restrict__ adj_s,
                           float* denom_ag, float* denom_s, float* kl) {
    __shared__ float red[256];
    int row = blockIdx.x, tid = threadIdx.x;
    float va = adj_ag[(i64)row * CL + tid];
    float vs = adj_s[(i64)row * CL + tid];
    float suma = blk_sum(va, red);
    float sums = blk_sum(vs, red);
    float mxa = blk_max(va, red);
    float mxs = blk_max(vs, red);
    float sea = blk_sum(__expf(va - mxa), red);
    float ses = blk_sum(__expf(vs - mxs), red);
    float lsea = mxa + __logf(sea);
    float lses = mxs + __logf(ses);
    float lq = vs - lses, lp = va - lsea;
    float tot = blk_sum(__expf(lq) * (lq - lp), red);
    if (tid == 0) {
        denom_ag[row] = suma + 1.f;
        denom_s[row]  = sums + 1.f;
        atomicAdd(kl, tot);
    }
}

// ---------------------------------------------------------------- row softmax (cols = 256) -> bf16
__global__ void k_softmax_rows(const float* __restrict__ in, u16* __restrict__ outb) {
    __shared__ float red[256];
    int row = blockIdx.x, tid = threadIdx.x;
    float v = in[(i64)row * CL + tid];
    float mx = blk_max(v, red);
    float e  = __expf(v - mx);
    float s  = blk_sum(e, red);
    outb[(i64)row * CL + tid] = f2bf(e / s);
}

// ---------------------------------------------------------------- final masked pool + classifier + kl_loss
__global__ void k_final(const float* __restrict__ outag, const float* __restrict__ outs,
                        const float* __restrict__ pooled, const float* __restrict__ amask,
                        const float* __restrict__ wc, const float* __restrict__ bc,
                        const float* __restrict__ kl, float* __restrict__ out) {
    __shared__ float cat[2 * CD];
    __shared__ float red[256];
    int b = blockIdx.x, tid = threadIdx.x;
    float wn = blk_sum(amask[b * CL + tid], red);
    for (int d = tid; d < CMEM; d += 256) {
        float s1 = 0.f, s2 = 0.f;
        for (int l = 0; l < CL; ++l) {
            float m = amask[b * CL + l];
            if (m != 0.f) {
                s1 += outag[((i64)b * CL + l) * CMEM + d] * m;
                s2 += outs[((i64)b * CL + l) * CMEM + d] * m;
            }
        }
        cat[d]        = s1 / wn;
        cat[CMEM + d] = s2 / wn;
    }
    for (int d = tid; d < CD; d += 256) cat[2 * CMEM + d] = pooled[b * CD + d];
    __syncthreads();
    for (int p = 0; p < CPOL; ++p) {
        float part = 0.f;
        for (int i = tid; i < 2 * CD; i += 256) part += cat[i] * wc[i * CPOL + p];
        float tot = blk_sum(part, red);
        if (tid == 0) out[b * CPOL + p] = tot + bc[p];
    }
    if (b == 0 && tid == 0) out[CB * CPOL] = __expf(-0.1f * kl[0]);   // GAMA=0.1
}

// ================================================================ host
extern "C" void kernel_launch(void* const* d_in, const int* in_sizes, int n_in,
                              void* d_out, int out_size, void* d_ws, size_t ws_size,
                              hipStream_t stream) {
    (void)in_sizes; (void)n_in; (void)out_size; (void)ws_size;
    const float* seq    = (const float*)d_in[0];
    const float* pooled = (const float*)d_in[1];
    const float* adjr   = (const float*)d_in[2];
    const float* smask  = (const float*)d_in[3];
    const float* amask  = (const float*)d_in[4];
    const float* ln_g   = (const float*)d_in[5];
    const float* ln_b   = (const float*)d_in[6];
    const float* wq     = (const float*)d_in[7];
    const float* bq     = (const float*)d_in[8];
    const float* wk     = (const float*)d_in[9];
    const float* bk     = (const float*)d_in[10];
    const float* wd     = (const float*)d_in[11];
    const float* bd     = (const float*)d_in[12];
    const float* wm     = (const float*)d_in[13];
    const float* bm     = (const float*)d_in[14];
    const float* wa0    = (const float*)d_in[15];
    const float* ba0    = (const float*)d_in[16];
    const float* wa1    = (const float*)d_in[17];
    const float* ba1    = (const float*)d_in[18];
    const float* wsl0   = (const float*)d_in[19];
    const float* bs0    = (const float*)d_in[20];
    const float* wsl1   = (const float*)d_in[21];
    const float* bs1    = (const float*)d_in[22];
    const float* af1    = (const float*)d_in[23];
    const float* af2    = (const float*)d_in[24];
    const float* wc     = (const float*)d_in[25];
    const float* bc     = (const float*)d_in[26];
    float* out = (float*)d_out;

    char* base = (char*)d_ws;
    size_t off = 0;
    auto alloc = [&](size_t bytes) -> void* {
        void* p = base + off;
        off = (off + bytes + 255) & ~(size_t)255;
        return p;
    };

    // ---- persistent region
    float* xf       = (float*)alloc(sizeof(float) * CB * CL * CD);
    float* adj_s_f  = (float*)alloc(sizeof(float) * CB * CL * CL);
    float* adj_ag_f = (float*)alloc(sizeof(float) * CB * CL * CL);
    u16*   adj_s_b  = (u16*)alloc(sizeof(u16) * CB * CL * CL);
    u16*   adj_ag_b = (u16*)alloc(sizeof(u16) * CB * CL * CL);
    float* denom_ag = (float*)alloc(sizeof(float) * CB * CL);
    float* denom_s  = (float*)alloc(sizeof(float) * CB * CL);
    float* klbuf    = (float*)alloc(sizeof(float) * 64);
    u16* wa0T = (u16*)alloc(sizeof(u16) * CMEM * CD);
    u16* wa1T = (u16*)alloc(sizeof(u16) * CMEM * CMEM);
    u16* ws0T = (u16*)alloc(sizeof(u16) * CMEM * CD);
    u16* ws1T = (u16*)alloc(sizeof(u16) * CMEM * CMEM);
    u16* af1T = (u16*)alloc(sizeof(u16) * CMEM * CMEM);
    u16* af2T = (u16*)alloc(sizeof(u16) * CMEM * CMEM);

    size_t mark = off;
    // ---- stage 1 scratch (attention build)
    u16* xb     = (u16*)alloc(sizeof(u16) * CB * CL * CD);
    u16* qb     = (u16*)alloc(sizeof(u16) * CB * CL * CD);
    u16* kb     = (u16*)alloc(sizeof(u16) * CB * CL * CD);
    u16* wqT    = (u16*)alloc(sizeof(u16) * CD * CD);
    u16* wkT    = (u16*)alloc(sizeof(u16) * CD * CD);
    u16* wdT    = (u16*)alloc(sizeof(u16) * CDK * CD);
    u16* wmT    = (u16*)alloc(sizeof(u16) * CH * CDK * CDK);
    float* asp_t = (float*)alloc(sizeof(float) * CB * CL * CDK);
    u16* asp_b  = (u16*)alloc(sizeof(u16) * CB * CL * CDK);
    u16* aspm_b = (u16*)alloc(sizeof(u16) * CB * CH * CL * CDK);
    float* asa  = (float*)alloc(sizeof(float) * CB * CL * CL);
    float* pre  = (float*)alloc(sizeof(float) * CB * CL * CL);

    // ---- stage 2 scratch (GCN) reuses the stage-1 region (stage 1 fully consumed first)
    off = mark;
    u16* outT_ag = (u16*)alloc(sizeof(u16) * CB * CD * CL);
    u16* outT_s  = (u16*)alloc(sizeof(u16) * CB * CD * CL);
    u16* t_ag    = (u16*)alloc(sizeof(u16) * CB * CL * CD);
    u16* t_s     = (u16*)alloc(sizeof(u16) * CB * CL * CD);
    float* g_ag_f = (float*)alloc(sizeof(float) * CB * CL * CMEM);
    float* g_s_f  = (float*)alloc(sizeof(float) * CB * CL * CMEM);
    u16* g_ag_b  = (u16*)alloc(sizeof(u16) * CB * CL * CMEM);
    u16* g_s_b   = (u16*)alloc(sizeof(u16) * CB * CL * CMEM);
    u16* u_ag_b  = (u16*)alloc(sizeof(u16) * CB * CL * CMEM);
    u16* u_s_b   = (u16*)alloc(sizeof(u16) * CB * CL * CMEM);
    float* sc1_f = (float*)alloc(sizeof(float) * CB * CL * CL);
    float* sc2_f = (float*)alloc(sizeof(float) * CB * CL * CL);
    u16* A1_b    = (u16*)alloc(sizeof(u16) * CB * CL * CL);
    u16* A2_b    = (u16*)alloc(sizeof(u16) * CB * CL * CL);
    u16* gT_ag   = (u16*)alloc(sizeof(u16) * CB * CMEM * CL);
    u16* gT_s    = (u16*)alloc(sizeof(u16) * CB * CMEM * CL);
    float* oA_ag = (float*)alloc(sizeof(float) * CB * CL * CMEM);
    float* oA_s  = (float*)alloc(sizeof(float) * CB * CL * CMEM);

    auto gemm = [&](const u16* A, i64 sA, int aDiv, int aMod,
                    const u16* Bt, i64 sB, int bDiv, int bMod,
                    float* Cf, u16* Cb, i64 sC,
                    int M, int N, int K, int lda, int ldb, int ldc, int batch,
                    const float* bias, const float* rowdiv, i64 rdStride, int relu) {
        dim3 g(N / 64, M / 64, batch);   // 16x64 strip per wave, 4 waves per block
        k_gemm_nt<<<g, 128, 0, stream>>>(A, sA, aDiv, aMod, Bt, sB, bDiv, bMod,
                                         Cf, Cb, sC, M, N, K, lda, ldb, ldc,
                                         bias, rowdiv, rdStride, relu);
    };
    auto tr = [&](const float* in, u16* o, int R, int C, int batch) {
        dim3 g((R * C + 255) / 256, 1, batch);
        k_transpose<<<g, 256, 0, stream>>>(in, o, R, C);
    };

    // ---- zero accumulators (poisoned workspace; also needed per graph replay)
    k_zero<<<512, 256, 0, stream>>>(pre, (i64)CB * CL * CL);
    k_zero<<<512, 256, 0, stream>>>(asa, (i64)CB * CL * CL);
    k_zero<<<1, 64, 0, stream>>>(klbuf, 64);

    // ---- weight prep: transpose + bf16 so every GEMM is NT
    tr(wq, wqT, CD, CD, 1);
    tr(wk, wkT, CD, CD, 1);
    tr(wd, wdT, CD, CDK, 1);
    tr(wm, wmT, CDK, CDK, CH);
    tr(wa0, wa0T, CD, CMEM, 1);
    tr(wa1, wa1T, CMEM, CMEM, 1);
    tr(wsl0, ws0T, CD, CMEM, 1);
    tr(wsl1, ws1T, CMEM, CMEM, 1);
    tr(af1, af1T, CMEM, CMEM, 1);
    tr(af2, af2T, CMEM, CMEM, 1);

    // ---- LayerNorm
    k_layernorm<<<CB * CL, 256, 0, stream>>>(seq, ln_g, ln_b, xf, xb);

    // ---- q = x@wq+bq, k = x@wk+bk (bf16 outputs)
    gemm(xb, 0, 1, BIGMOD, wqT, 0, 1, BIGMOD, nullptr, qb, 0,
         CB * CL, CD, CD, CD, CD, CD, 1, bq, nullptr, 0, 0);
    gemm(xb, 0, 1, BIGMOD, wkT, 0, 1, BIGMOD, nullptr, kb, 0,
         CB * CL, CD, CD, CD, CD, CD, 1, bk, nullptr, 0, 0);

    // ---- asp = aspect_mask*(x@wd) + bd
    gemm(xb, 0, 1, BIGMOD, wdT, 0, 1, BIGMOD, asp_t, nullptr, 0,
         CB * CL, CDK, CD, CD, CD, CDK, 1, nullptr, nullptr, 0, 0);
    k_asp_fix<<<(CB * CL * CDK) / 256, 256, 0, stream>>>(asp_t, amask, bd, asp_b);

    // ---- aspm[b,h] = asp[b] @ weight_m[h]   (batched, A shared over h, B over b)
    gemm(asp_b, (i64)CL * CDK, CH, CB, wmT, (i64)CDK * CDK, 1, CH,
         nullptr, aspm_b, (i64)CL * CDK,
         CL, CDK, CDK, CDK, CDK, CDK, CB * CH, nullptr, nullptr, 0, 0);

    // ---- fused: asps/tanh/avg/aspect_scores -> asa ; scores/softmax -> pre
    size_t ldsAsps = (size_t)(256 * LDW + 256) * 4;
    k_asps_fused<<<CB * CH, 256, ldsAsps, stream>>>(aspm_b, kb, amask, bm, asa);
    size_t ldsAttn = (size_t)(256 * LDW) * 4;
    k_attn_fused<<<CB * CH, 256, ldsAttn, stream>>>(qb, kb, smask, pre);

    // ---- adjacency matrices + denominators + KL
    k_adj_build<<<(CB * CL * CL) / 256, 256, 0, stream>>>(pre, asa, adjr, smask,
                                                          adj_s_f, adj_s_b, adj_ag_f, adj_ag_b);
    k_rowstats<<<CB * CL, 256, 0, stream>>>(adj_ag_f, adj_s_f, denom_ag, denom_s, klbuf);

    // ---- dual GCN, 2 layers
    for (int li = 0; li < 2; ++li) {
        int Din = (li == 0) ? CD : CMEM;
        const float* srcAg = (li == 0) ? xf : oA_ag;
        const float* srcS  = (li == 0) ? xf : oA_s;
        const u16* WaT = (li == 0) ? wa0T : wa1T;
        const u16* WsT = (li == 0) ? ws0T : ws1T;
        const float* baA = (li == 0) ? ba0 : ba1;
        const float* bsA = (li == 0) ? bs0 : bs1;

        tr(srcAg, outT_ag, CL, Din, CB);
        tr(srcS,  outT_s,  CL, Din, CB);

        // t = adj @ out        (batched over b)
        gemm(adj_ag_b, (i64)CL * CL, 1, BIGMOD, outT_ag, (i64)Din * CL, 1, BIGMOD,
             nullptr, t_ag, (i64)CL * Din, CL, Din, CL, CL, CL, Din, CB, nullptr, nullptr, 0, 0);
        gemm(adj_s_b, (i64)CL * CL, 1, BIGMOD, outT_s, (i64)Din * CL, 1, BIGMOD,
             nullptr, t_s, (i64)CL * Din, CL, Din, CL, CL, CL, Din, CB, nullptr, nullptr, 0, 0);

        // g = relu((t@W + b)/denom)
        gemm(t_ag, 0, 1, BIGMOD, WaT, 0, 1, BIGMOD, g_ag_f, g_ag_b, 0,
             CB * CL, CMEM, Din, Din, Din, CMEM, 1, baA, denom_ag, 0, 1);
        gemm(t_s, 0, 1, BIGMOD, WsT, 0, 1, BIGMOD, g_s_f, g_s_b, 0,
             CB * CL, CMEM, Din, Din, Din, CMEM, 1, bsA, denom_s, 0, 1);

        // u = g @ affine
        gemm(g_ag_b, 0, 1, BIGMOD, af1T, 0, 1, BIGMOD, nullptr, u_ag_b, 0,
             CB * CL, CMEM, CMEM, CMEM, CMEM, CMEM, 1, nullptr, nullptr, 0, 0);
        gemm(g_s_b, 0, 1, BIGMOD, af2T, 0, 1, BIGMOD, nullptr, u_s_b, 0,
             CB * CL, CMEM, CMEM, CMEM, CMEM, CMEM, 1, nullptr, nullptr, 0, 0);

        // scores = u @ g^T (NT directly; contraction over features)
        gemm(u_ag_b, (i64)CL * CMEM, 1, BIGMOD, g_s_b, (i64)CL * CMEM, 1, BIGMOD,
             sc1_f, nullptr, (i64)CL * CL, CL, CL, CMEM, CMEM, CMEM, CL, CB, nullptr, nullptr, 0, 0);
        gemm(u_s_b, (i64)CL * CMEM, 1, BIGMOD, g_ag_b, (i64)CL * CMEM, 1, BIGMOD,
             sc2_f, nullptr, (i64)CL * CL, CL, CL, CMEM, CMEM, CMEM, CL, CB, nullptr, nullptr, 0, 0);

        k_softmax_rows<<<CB * CL, 256, 0, stream>>>(sc1_f, A1_b);
        k_softmax_rows<<<CB * CL, 256, 0, stream>>>(sc2_f, A2_b);

        tr(g_s_f, gT_s, CL, CMEM, CB);
        tr(g_ag_f, gT_ag, CL, CMEM, CB);

        // out_ag = A1 @ g_s ; out_s = A2 @ g_ag
        gemm(A1_b, (i64)CL * CL, 1, BIGMOD, gT_s, (i64)CMEM * CL, 1, BIGMOD,
             oA_ag, nullptr, (i64)CL * CMEM, CL, CMEM, CL, CL, CL, CMEM, CB, nullptr, nullptr, 0, 0);
        gemm(A2_b, (i64)CL * CL, 1, BIGMOD, gT_ag, (i64)CMEM * CL, 1, BIGMOD,
             oA_s, nullptr, (i64)CL * CMEM, CL, CMEM, CL, CL, CL, CMEM, CB, nullptr, nullptr, 0, 0);
    }

    // ---- final pooled classifier + kl_loss
    k_final<<<CB, 256, 0, stream>>>(oA_ag, oA_s, pooled, amask, wc, bc, klbuf, out);
}
